// TFN_51453708206529
// MI455X (gfx1250) — compile-verified, matrix operand order checked by reading, not run on hardware
//
#include <hip/hip_runtime.h>
#include <hip/hip_bf16.h>
#include <math.h>

typedef __attribute__((ext_vector_type(16))) _Float16 v16h;
typedef __attribute__((ext_vector_type(8)))  _Float16 v8h;
typedef __attribute__((ext_vector_type(8)))  float    v8f;

#define B_ROWS   8192
#define KPAD     1120      // 1089 padded to 35 chunks of 32
#define NPAD     1120      // 1089 padded to 70 tiles of 16
#define NCHUNK   35
#define NTILES   70

union V16 { v16h v; v8h h[2]; };

#define WMMA_F16(A, B, C) \
    __builtin_amdgcn_wmma_f32_16x16x32_f16(false, (A).v, false, (B).v, (short)0, (C), false, false)

// ---------------------------------------------------------------------------
// Per-column mean and 1/sqrt(var+eps) over the batch (BatchNorm training stats)
// ---------------------------------------------------------------------------
__global__ void colstats_kernel(const float* __restrict__ x, int rows, int cols,
                                float* __restrict__ mu, float* __restrict__ rstd) {
    __shared__ float s1[256], s2[256];
    const int col = blockIdx.x;
    float a = 0.f, q = 0.f;
    for (int r = threadIdx.x; r < rows; r += blockDim.x) {
        float v = x[r * cols + col];
        a += v; q += v * v;
    }
    s1[threadIdx.x] = a; s2[threadIdx.x] = q;
    __syncthreads();
    for (int off = 128; off > 0; off >>= 1) {
        if ((int)threadIdx.x < off) {
            s1[threadIdx.x] += s1[threadIdx.x + off];
            s2[threadIdx.x] += s2[threadIdx.x + off];
        }
        __syncthreads();
    }
    if (threadIdx.x == 0) {
        float m = s1[0] / (float)rows;
        float v = s2[0] / (float)rows - m * m;
        mu[col]   = m;
        rstd[col] = rsqrtf(v + 1e-5f);
    }
}

// ---------------------------------------------------------------------------
// Normalized, padded, f16 activations: Xn[b,k] = BN(x[b,k]) for k<D else 0.
// ---------------------------------------------------------------------------
__global__ void xn_kernel(const float* __restrict__ x, int D, int Dpad,
                          const float* __restrict__ mu, const float* __restrict__ rstd,
                          const float* __restrict__ gamma, const float* __restrict__ beta,
                          _Float16* __restrict__ Xn) {
    const int idx = blockIdx.x * blockDim.x + threadIdx.x;
    if (idx >= B_ROWS * Dpad) return;
    const int b = idx / Dpad, k = idx % Dpad;
    float v = 0.f;
    if (k < D) v = (x[b * D + k] - mu[k]) * rstd[k] * gamma[k] + beta[k];
    Xn[idx] = (_Float16)v;
}

// ---------------------------------------------------------------------------
// Swizzle a [D,32] f32 weight into WMMA-B per-lane layout (f16), zero-padded.
// ---------------------------------------------------------------------------
__global__ void wswz_kernel(const float* __restrict__ W, int D, int nchunks,
                            _Float16* __restrict__ Wsw) {
    const int idx = blockIdx.x * blockDim.x + threadIdx.x;
    if (idx >= nchunks * 2 * 32 * 16) return;
    const int e = idx & 15;
    const int l = (idx >> 4) & 31;
    const int t = (idx >> 9) & 1;
    const int c = idx >> 10;
    const int k = c * 32 + ((l >> 4) << 3) + (e & 7) + ((e >> 3) << 4);
    const int n = t * 16 + (l & 15);
    Wsw[idx] = (_Float16)((k < D) ? W[k * 32 + n] : 0.f);
}

// ---------------------------------------------------------------------------
// Subnet MLP on WMMA with software-pipelined layer-1 K loop.
// One wave per 16 batch rows; inter-layer hand-off via WMMA-A-ordered LDS.
// ---------------------------------------------------------------------------
__device__ __forceinline__ void ld_sub(const _Float16* a, const _Float16* w,
                                       V16& ua, V16& ub0, V16& ub1) {
    ua.h[0]  = *(const v8h*)(a);
    ua.h[1]  = *(const v8h*)(a + 16);
    ub0.h[0] = *(const v8h*)(w);
    ub0.h[1] = *(const v8h*)(w + 8);
    ub1.h[0] = *(const v8h*)(w + 512);
    ub1.h[1] = *(const v8h*)(w + 520);
}

__global__ void subnet_kernel(const _Float16* __restrict__ Xn, int nchunks, int Dpad,
                              const _Float16* __restrict__ W1sw, const float* __restrict__ b1,
                              const _Float16* __restrict__ W2sw, const float* __restrict__ b2,
                              const _Float16* __restrict__ W3sw, const float* __restrict__ b3,
                              float* __restrict__ out1) {
    __shared__ __attribute__((aligned(32))) _Float16 abuf[32][16];
    const int l   = threadIdx.x;
    const int lm  = l & 15;
    const int hi  = l >> 4;
    const int rowBase = blockIdx.x * 16;

    // ---- layer 1: ping-pong pipelined over K chunks ----
    v8f c0 = {}, c1 = {};
    const _Float16* arow = Xn + (rowBase + lm) * Dpad + hi * 8;
    const _Float16* wrow = W1sw + l * 16;
    V16 aA, bA0, bA1, aB, bB0, bB1;
    ld_sub(arow, wrow, aA, bA0, bA1);
    for (int c = 0; c + 2 <= nchunks; c += 2) {
        ld_sub(arow + (c + 1) * 32, wrow + (c + 1) * 1024, aB, bB0, bB1);
        c0 = WMMA_F16(aA, bA0, c0);
        c1 = WMMA_F16(aA, bA1, c1);
        if (c + 2 < nchunks)
            ld_sub(arow + (c + 2) * 32, wrow + (c + 2) * 1024, aA, bA0, bA1);
        c0 = WMMA_F16(aB, bB0, c0);
        c1 = WMMA_F16(aB, bB1, c1);
    }
    if (nchunks & 1) {
        c0 = WMMA_F16(aA, bA0, c0);
        c1 = WMMA_F16(aA, bA1, c1);
    }
    // bias+relu, scatter into WMMA-A order
    {
        const int dl = 16 * ((lm >> 3) & 1);
        const int e0 = lm & 7;
#pragma unroll
        for (int r = 0; r < 8; ++r) {
            const int m = r + 8 * hi;
            abuf[m + dl][e0]     = (_Float16)fmaxf(c0[r] + b1[lm],      0.f);
            abuf[m + dl][e0 + 8] = (_Float16)fmaxf(c1[r] + b1[16 + lm], 0.f);
        }
    }
    __syncthreads();

    // ---- layer 2 ----
    v8f d0 = {}, d1 = {};
    {
        V16 ua, ub0, ub1;
        ua.h[0] = *(const v8h*)&abuf[l][0];
        ua.h[1] = *(const v8h*)&abuf[l][8];
        ub0.h[0] = *(const v8h*)(W2sw + l * 16);
        ub0.h[1] = *(const v8h*)(W2sw + l * 16 + 8);
        ub1.h[0] = *(const v8h*)(W2sw + 512 + l * 16);
        ub1.h[1] = *(const v8h*)(W2sw + 520 + l * 16);
        d0 = WMMA_F16(ua, ub0, d0);
        d1 = WMMA_F16(ua, ub1, d1);
    }
    __syncthreads();
    {
        const int dl = 16 * ((lm >> 3) & 1);
        const int e0 = lm & 7;
#pragma unroll
        for (int r = 0; r < 8; ++r) {
            const int m = r + 8 * hi;
            abuf[m + dl][e0]     = (_Float16)fmaxf(d0[r] + b2[lm],      0.f);
            abuf[m + dl][e0 + 8] = (_Float16)fmaxf(d1[r] + b2[16 + lm], 0.f);
        }
    }
    __syncthreads();

    // ---- layer 3 (results straight to global) ----
    v8f g0 = {}, g1 = {};
    {
        V16 ua, ub0, ub1;
        ua.h[0] = *(const v8h*)&abuf[l][0];
        ua.h[1] = *(const v8h*)&abuf[l][8];
        ub0.h[0] = *(const v8h*)(W3sw + l * 16);
        ub0.h[1] = *(const v8h*)(W3sw + l * 16 + 8);
        ub1.h[0] = *(const v8h*)(W3sw + 512 + l * 16);
        ub1.h[1] = *(const v8h*)(W3sw + 520 + l * 16);
        g0 = WMMA_F16(ua, ub0, g0);
        g1 = WMMA_F16(ua, ub1, g1);
    }
#pragma unroll
    for (int r = 0; r < 8; ++r) {
        const int m = rowBase + r + 8 * hi;
        out1[m * 33 + 1 + lm]  = fmaxf(g0[r] + b3[lm],      0.f);
        out1[m * 33 + 17 + lm] = fmaxf(g1[r] + b3[16 + lm], 0.f);
    }
    if (l < 16) out1[(rowBase + l) * 33] = 1.0f;
}

// ---------------------------------------------------------------------------
// A = cdus: A[b, j*33+i] = cd1[b,j]*us1[b,i] in f16, K padded to 1120 with 0.
// ---------------------------------------------------------------------------
__global__ void fuse_a_kernel(const float* __restrict__ cd1, const float* __restrict__ us1,
                              _Float16* __restrict__ A) {
    const int idx = blockIdx.x * blockDim.x + threadIdx.x;
    if (idx >= B_ROWS * KPAD) return;
    const int b = idx / KPAD, n = idx % KPAD;
    float v = 0.f;
    if (n < 1089) v = cd1[b * 33 + n / 33] * us1[b * 33 + n % 33];
    A[idx] = (_Float16)v;
}

// ---------------------------------------------------------------------------
// Pre-swizzled Wf1: Bsw[c][tile][lane][e] = Wf1r[c*32 + koff(l,e), tile*16 + (l&15)]
// where Wf1r[m, ko*33+o] = Wf1[(ko*1089+m)*33 + o]; zero pad outside 1089.
// ---------------------------------------------------------------------------
__global__ void bsw_kernel(const float* __restrict__ Wf1, _Float16* __restrict__ Bsw) {
    const int idx = blockIdx.x * blockDim.x + threadIdx.x;
    if (idx >= NCHUNK * NTILES * 32 * 16) return;
    const int e = idx & 15;
    const int l = (idx >> 4) & 31;
    const int t = (idx >> 9) % NTILES;
    const int c = (idx >> 9) / NTILES;
    const int k = c * 32 + ((l >> 4) << 3) + (e & 7) + ((e >> 3) << 4);
    const int n = t * 16 + (l & 15);
    float v = 0.f;
    if (k < 1089 && n < 1089) {
        const int ko = n / 33, o = n % 33;
        v = Wf1[(ko * 1089 + k) * 33 + o];
    }
    Bsw[idx] = (_Float16)v;
}

// ---------------------------------------------------------------------------
// Big GEMM: P[8192,1120] = A[8192,1120] @ B[1120,1120] (f16 in, f32 out).
// Wave tile = 32M x 32N, block = 4 waves = 128 rows x 32 cols, grid (64, 35).
// K loop software-pipelined 2 deep (ping-pong fragment registers) so WMMAs
// overlap next-chunk loads instead of s_wait_loadcnt 0 on every step.
// ---------------------------------------------------------------------------
__device__ __forceinline__ void ld_gemm(const _Float16* a0, const _Float16* a1,
                                        const _Float16* bb,
                                        V16& ua0, V16& ua1, V16& ub0, V16& ub1) {
    ua0.h[0] = *(const v8h*)(a0);
    ua0.h[1] = *(const v8h*)(a0 + 16);
    ua1.h[0] = *(const v8h*)(a1);
    ua1.h[1] = *(const v8h*)(a1 + 16);
    ub0.h[0] = *(const v8h*)(bb);
    ub0.h[1] = *(const v8h*)(bb + 8);
    ub1.h[0] = *(const v8h*)(bb + 512);
    ub1.h[1] = *(const v8h*)(bb + 520);
}

__global__ void gemm_kernel(const _Float16* __restrict__ A,
                            const _Float16* __restrict__ Bsw,
                            float* __restrict__ P) {
    const int tid = threadIdx.x;
    const int l   = tid & 31;
    const int w   = tid >> 5;
    const int lm  = l & 15;
    const int hi  = l >> 4;
    const int rowBase = blockIdx.x * 128 + w * 32;
    const int tile0   = blockIdx.y * 2;
    const int BSTEP   = NTILES * 512;            // halves per K-chunk in Bsw

    v8f c00 = {}, c01 = {}, c10 = {}, c11 = {};
    const _Float16* arow0 = A + (rowBase + lm) * KPAD + hi * 8;
    const _Float16* arow1 = arow0 + 16 * KPAD;
    const _Float16* bbase = Bsw + (tile0 * 32 + l) * 16;

    V16 aA0, aA1, bA0, bA1, aB0, aB1, bB0, bB1;
    ld_gemm(arow0, arow1, bbase, aA0, aA1, bA0, bA1);
    for (int c = 0; c + 2 <= NCHUNK; c += 2) {
        __builtin_prefetch(arow0 + (c + 8) * 32, 0, 1);
        __builtin_prefetch(arow1 + (c + 8) * 32, 0, 1);
        ld_gemm(arow0 + (c + 1) * 32, arow1 + (c + 1) * 32,
                bbase + (size_t)(c + 1) * BSTEP, aB0, aB1, bB0, bB1);
        c00 = WMMA_F16(aA0, bA0, c00);
        c01 = WMMA_F16(aA0, bA1, c01);
        c10 = WMMA_F16(aA1, bA0, c10);
        c11 = WMMA_F16(aA1, bA1, c11);
        if (c + 2 < NCHUNK)
            ld_gemm(arow0 + (c + 2) * 32, arow1 + (c + 2) * 32,
                    bbase + (size_t)(c + 2) * BSTEP, aA0, aA1, bA0, bA1);
        c00 = WMMA_F16(aB0, bB0, c00);
        c01 = WMMA_F16(aB0, bB1, c01);
        c10 = WMMA_F16(aB1, bB0, c10);
        c11 = WMMA_F16(aB1, bB1, c11);
    }
    // NCHUNK = 35 is odd: last chunk sits in the A set
    c00 = WMMA_F16(aA0, bA0, c00);
    c01 = WMMA_F16(aA0, bA1, c01);
    c10 = WMMA_F16(aA1, bA0, c10);
    c11 = WMMA_F16(aA1, bA1, c11);

    const int col0 = tile0 * 16 + lm;
#pragma unroll
    for (int r = 0; r < 8; ++r) {
        const int m0 = rowBase + r + 8 * hi;
        P[m0 * NPAD + col0]             = c00[r];
        P[m0 * NPAD + col0 + 16]        = c01[r];
        P[(m0 + 16) * NPAD + col0]      = c10[r];
        P[(m0 + 16) * NPAD + col0 + 16] = c11[r];
    }
}

// ---------------------------------------------------------------------------
// Epilogue: thread = (row, o).  h[o] = relu(bf1[o] + sum_k mk1[b,k]*P[b,k*33+o]);
// out = sigmoid(h@Wf2+bf2).  Lanes sweep o => P reads are contiguous per k.
// Block = 264 threads = 8 rows x 33 cols.
// ---------------------------------------------------------------------------
__global__ void epilogue_kernel(const float* __restrict__ P, const float* __restrict__ mk1,
                                const float* __restrict__ bf1, const float* __restrict__ Wf2,
                                const float* __restrict__ bf2, float* __restrict__ out) {
    __shared__ float hsh[8][33];
    const int o  = threadIdx.x % 33;
    const int rr = threadIdx.x / 33;          // 0..7
    const int b  = blockIdx.x * 8 + rr;
    const float* prow = P + b * NPAD;
    const float* mrow = mk1 + b * 33;
    float acc = bf1[o];
#pragma unroll 3
    for (int k = 0; k < 33; ++k) acc += mrow[k] * prow[k * 33 + o];
    hsh[rr][o] = fmaxf(acc, 0.f);
    __syncthreads();
    if (o < 5) {
        float z = bf2[o];
#pragma unroll
        for (int q = 0; q < 33; ++q) z += hsh[rr][q] * Wf2[q * 5 + o];
        out[b * 5 + o] = 1.f / (1.f + expf(-z));
    }
}

// ---------------------------------------------------------------------------
extern "C" void kernel_launch(void* const* d_in, const int* in_sizes, int n_in,
                              void* d_out, int out_size, void* d_ws, size_t ws_size,
                              hipStream_t stream) {
    (void)in_sizes; (void)n_in; (void)out_size; (void)ws_size;
    const float* US_x   = (const float*)d_in[0];
    const float* CDFI_x = (const float*)d_in[1];
    const float* mk_x   = (const float*)d_in[2];
    const float* us_p[8]; for (int i = 0; i < 8; ++i) us_p[i] = (const float*)d_in[3 + i];
    const float* cd_p[8]; for (int i = 0; i < 8; ++i) cd_p[i] = (const float*)d_in[11 + i];
    const float* mk_p[8]; for (int i = 0; i < 8; ++i) mk_p[i] = (const float*)d_in[19 + i];
    const float* Wf1 = (const float*)d_in[27];
    const float* bf1 = (const float*)d_in[28];
    const float* Wf2 = (const float*)d_in[29];
    const float* bf2 = (const float*)d_in[30];
    float* out = (float*)d_out;

    char* ws = (char*)d_ws;
    const size_t MB = 1024u * 1024u;
    float*    mu_us   = (float*)(ws + 0);
    float*    rstd_us = (float*)(ws + 4096);
    float*    mu_cd   = (float*)(ws + 8192);
    float*    rstd_cd = (float*)(ws + 12288);
    float*    mu_mk   = (float*)(ws + 16384);
    float*    rstd_mk = (float*)(ws + 20480);
    float*    us1     = (float*)(ws + 2 * MB);
    float*    cd1     = (float*)(ws + 4 * MB);
    float*    mk1     = (float*)(ws + 6 * MB);
    _Float16* Wsw_us  = (_Float16*)(ws + 7 * MB);
    _Float16* Wsw_cd  = (_Float16*)(ws + 7 * MB + 128 * 1024);
    _Float16* Wsw_mk  = (_Float16*)(ws + 7 * MB + 256 * 1024);
    _Float16* Ap      = (_Float16*)(ws + 8 * MB);             // 18.4 MB
    _Float16* Bsw     = (_Float16*)(ws + 28 * MB);            // 2.51 MB
    _Float16* Xn_us   = (_Float16*)(ws + 32 * MB);            // dead after subnets
    _Float16* Xn_cd   = (_Float16*)(ws + 48 * MB);            // dead after subnets
    _Float16* Xn_mk   = (_Float16*)(ws + 64 * MB);            // dead after subnets
    float*    P       = (float*)(ws + 32 * MB);               // overlaps Xn (ordered)

    const int W1OFF = 96 * 1024 / 2;   // halves
    const int W3OFF = 100 * 1024 / 2;

    colstats_kernel<<<1024, 256, 0, stream>>>(US_x,   B_ROWS, 1024, mu_us, rstd_us);
    colstats_kernel<<<1024, 256, 0, stream>>>(CDFI_x, B_ROWS, 1024, mu_cd, rstd_cd);
    colstats_kernel<<<10,   256, 0, stream>>>(mk_x,   B_ROWS, 10,   mu_mk, rstd_mk);

    xn_kernel<<<(B_ROWS * 1024 + 255) / 256, 256, 0, stream>>>(US_x,   1024, 1024, mu_us, rstd_us, us_p[0], us_p[1], Xn_us);
    xn_kernel<<<(B_ROWS * 1024 + 255) / 256, 256, 0, stream>>>(CDFI_x, 1024, 1024, mu_cd, rstd_cd, cd_p[0], cd_p[1], Xn_cd);
    xn_kernel<<<(B_ROWS * 32   + 255) / 256, 256, 0, stream>>>(mk_x,   10,   32,   mu_mk, rstd_mk, mk_p[0], mk_p[1], Xn_mk);

    wswz_kernel<<<(32 * 1024 + 255) / 256, 256, 0, stream>>>(us_p[2], 1024, 32, Wsw_us);
    wswz_kernel<<<4, 256, 0, stream>>>(us_p[4], 32, 1, Wsw_us + W1OFF);
    wswz_kernel<<<4, 256, 0, stream>>>(us_p[6], 32, 1, Wsw_us + W3OFF);
    wswz_kernel<<<(32 * 1024 + 255) / 256, 256, 0, stream>>>(cd_p[2], 1024, 32, Wsw_cd);
    wswz_kernel<<<4, 256, 0, stream>>>(cd_p[4], 32, 1, Wsw_cd + W1OFF);
    wswz_kernel<<<4, 256, 0, stream>>>(cd_p[6], 32, 1, Wsw_cd + W3OFF);
    wswz_kernel<<<4, 256, 0, stream>>>(mk_p[2], 10, 1, Wsw_mk);
    wswz_kernel<<<4, 256, 0, stream>>>(mk_p[4], 32, 1, Wsw_mk + W1OFF);
    wswz_kernel<<<4, 256, 0, stream>>>(mk_p[6], 32, 1, Wsw_mk + W3OFF);

    subnet_kernel<<<B_ROWS / 16, 32, 0, stream>>>(Xn_us, 32, 1024,
        Wsw_us, us_p[3], Wsw_us + W1OFF, us_p[5], Wsw_us + W3OFF, us_p[7], us1);
    subnet_kernel<<<B_ROWS / 16, 32, 0, stream>>>(Xn_cd, 32, 1024,
        Wsw_cd, cd_p[3], Wsw_cd + W1OFF, cd_p[5], Wsw_cd + W3OFF, cd_p[7], cd1);
    subnet_kernel<<<B_ROWS / 16, 32, 0, stream>>>(Xn_mk, 1, 32,
        Wsw_mk, mk_p[3], Wsw_mk + W1OFF, mk_p[5], Wsw_mk + W3OFF, mk_p[7], mk1);

    fuse_a_kernel<<<(B_ROWS * KPAD + 255) / 256, 256, 0, stream>>>(cd1, us1, Ap);
    bsw_kernel<<<(NCHUNK * NTILES * 32 * 16 + 255) / 256, 256, 0, stream>>>(Wf1, Bsw);

    gemm_kernel<<<dim3(B_ROWS / 128, NPAD / 32), 128, 0, stream>>>(Ap, Bsw, P);

    epilogue_kernel<<<B_ROWS / 8, 264, 0, stream>>>(P, mk1, bf1, Wf2, bf2, out);
}
